// Attention_76682346102812
// MI455X (gfx1250) — matrix operand order repaired and checked by reading, not compiled, not run on hardware
//
#include <hip/hip_runtime.h>
#include <hip/hip_bf16.h>
#include <math.h>
#include <stdint.h>

#define NB 4
#define NN 4096
#define CC 192
#define HDIM 64
#define ATT_SCALE 0.07216878364870323f   // 192^-0.5
#define LN_EPS 1e-5f

typedef _Float16 h16 __attribute__((ext_vector_type(16)));
typedef _Float16 h8  __attribute__((ext_vector_type(8)));
typedef float f8 __attribute__((ext_vector_type(8)));

__device__ __forceinline__ f8 f8_zero() {
  f8 a;
#pragma unroll
  for (int i = 0; i < 8; ++i) a[i] = 0.0f;
  return a;
}

__device__ __forceinline__ h16 join16(h8 lo, h8 hi) {
  return __builtin_shufflevector(lo, hi, 0, 1, 2, 3, 4, 5, 6, 7,
                                 8, 9, 10, 11, 12, 13, 14, 15);
}

// CDNA5 async memory->LDS copy, 16B per lane, tracked by ASYNCcnt.
// lds_off = DS-space byte address (low 32 bits of generic LDS pointer, ISA 10.2).
__device__ __forceinline__ void async_ld_b128(uint32_t lds_off, const void* gaddr) {
  asm volatile("global_load_async_to_lds_b128 %0, %1, off"
               :: "v"(lds_off), "v"(gaddr) : "memory");
}
#define WAIT_ASYNC(n) asm volatile("s_wait_asynccnt " #n ::: "memory")

__device__ __forceinline__ uint32_t lds_addr(const void* p) {
  return (uint32_t)(uintptr_t)p;
}

// A fragment: 16x32 (MxK) f16 from row-major source, leading dim ld (halves).
// lane<16 holds K {0..15}, lane>=16 holds K {16..31}; row = lane%16.
__device__ __forceinline__ h16 load_a_frag(const _Float16* base, int ld) {
  const int lane = threadIdx.x & 31;
  const _Float16* p = base + (lane & 15) * ld + ((lane >> 4) << 4);
  return join16(*(const h8*)p, *(const h8*)(p + 8));
}

// B fragment: 32x16 (KxN) f16 from K-CONTIGUOUS (column-major) source:
// column n lives at base + n*ld, K contiguous. col = lane%16, elem i -> K (lane/16)*16+i.
__device__ __forceinline__ h16 load_b_fragT(const _Float16* base, int ld) {
  const int lane = threadIdx.x & 31;
  const _Float16* p = base + (lane & 15) * ld + ((lane >> 4) << 4);
  return join16(*(const h8*)p, *(const h8*)(p + 8));
}

// ---------------- conversion / packing kernels ----------------

__global__ void k_cvt_x(const float* __restrict__ X, _Float16* __restrict__ Xh,
                        _Float16* __restrict__ Ximg) {
  int idx = blockIdx.x * 256 + threadIdx.x;
  int b = idx / (NN * CC);
  int rem = idx - b * NN * CC;
  int n = rem / CC;
  int c = rem - n * CC;
  float v = X[idx];
  Xh[idx] = (_Float16)v;
  Ximg[((size_t)b * CC + c) * NN + n] = (_Float16)v;   // (B,C,H*W) layout
}

// plain f32 -> f16 convert (weights keep native N x K layout == B^T, K-contiguous)
__global__ void k_f2h(const float* __restrict__ S, _Float16* __restrict__ D) {
  int idx = blockIdx.x * 256 + threadIdx.x;
  D[idx] = (_Float16)S[idx];
}

// ---------------- generic WMMA GEMM: C[MxN] = A[MxK] * BT^T + bias ----------------
// BT is N x K (K-contiguous). 256 thr = 8 waves; tile 128x64, K-step 32.
// Double-buffered LDS fed by global_load_async_to_lds_b128 (ASYNCcnt pipeline):
// tile i+1 streams into buf^1 while WMMAs consume buf.

__global__ __launch_bounds__(256) void k_gemm(
    const _Float16* __restrict__ A, const _Float16* __restrict__ BT,
    const float* __restrict__ bias, float* __restrict__ Cf,
    _Float16* __restrict__ Ch, int M, int K, int Nld) {
  __shared__ __align__(16) _Float16 As[2][128 * 32];   // [buf][row][k]
  __shared__ __align__(16) _Float16 Bs[2][64 * 32];    // [buf][col][k] (K-contig)
  const int tid = threadIdx.x;
  const int m0 = blockIdx.x * 128;
  const int n0 = blockIdx.y * 64;
  const int wave = tid >> 5;
  f8 acc[4];
#pragma unroll
  for (int j = 0; j < 4; ++j) acc[j] = f8_zero();

  const int pa0 = tid * 8, pa1 = tid * 8 + 2048, pb = tid * 8;
  int ga0 = m0 + (pa0 >> 5); if (ga0 >= M) ga0 = M - 1;
  int ga1 = m0 + (pa1 >> 5); if (ga1 >= M) ga1 = M - 1;
  const int ca0 = pa0 & 31, ca1 = pa1 & 31;
  const int nb = n0 + (pb >> 5), cb = pb & 31;

  auto issue = [&](int k0, int buf) {
    async_ld_b128(lds_addr(&As[buf][pa0]), &A[(size_t)ga0 * K + k0 + ca0]);
    async_ld_b128(lds_addr(&As[buf][pa1]), &A[(size_t)ga1 * K + k0 + ca1]);
    async_ld_b128(lds_addr(&Bs[buf][pb]), &BT[(size_t)nb * K + k0 + cb]);
  };

  const int nk = K / 32;
  issue(0, 0);
  for (int i = 0; i < nk; ++i) {
    const int buf = i & 1;
    if (i + 1 < nk) {
      issue((i + 1) * 32, buf ^ 1);   // overlap next tile with this tile's math
      WAIT_ASYNC(0x3);                // older tile (3 ops/wave) has landed
    } else {
      WAIT_ASYNC(0x0);
    }
    __syncthreads();
    h16 af = load_a_frag(&As[buf][(wave << 4) * 32], 32);
#pragma unroll
    for (int j = 0; j < 4; ++j) {
      h16 bf = load_b_fragT(&Bs[buf][(j * 16) * 32], 32);
      acc[j] = __builtin_amdgcn_wmma_f32_16x16x32_f16(false, af, false, bf,
                                                      (short)0, acc[j], false, false);
    }
    __syncthreads();   // reads done before buf is refilled in iter i+2
  }
  const int lane = tid & 31;
  const int cn = lane & 15;
  const int cm = (lane >> 4) << 3;
  const int rowbase = m0 + (wave << 4) + cm;
#pragma unroll
  for (int j = 0; j < 4; ++j) {
    int col = n0 + j * 16 + cn;
    float bv = bias[col];
#pragma unroll
    for (int r = 0; r < 8; ++r) {
      int row = rowbase + r;
      if (row < M) {
        float v = acc[j][r] + bv;
        if (Ch) Ch[(size_t)row * Nld + col] = (_Float16)v;
        else    Cf[(size_t)row * Nld + col] = v;
      }
    }
  }
}

// ---------------- implicit-GEMM stride-1 VALID conv (per batch) ----------------
// Out[b][m][192] = im2col(Ximg[b]) (m x 192*KK*KK) @ W^T + bias.
// W (f16 copy of sr_w) is (co, ci*KK*KK) = N x K, K-contiguous.
// B tile uses the async double-buffer; A tile is an im2col LDS gather.

template <int KK>
__global__ __launch_bounds__(256) void k_conv_gemm(
    const _Float16* __restrict__ Ximg, const _Float16* __restrict__ Wt,
    const float* __restrict__ bias, float* __restrict__ Out, int side, int m) {
  constexpr int KTOT = CC * KK * KK;
  __shared__ __align__(16) _Float16 As[128 * 32];
  __shared__ __align__(16) _Float16 Bs[2][64 * 32];
  const int tid = threadIdx.x;
  const int b = blockIdx.z;
  const int m0 = blockIdx.x * 128;
  const int n0 = blockIdx.y * 64;
  const int wave = tid >> 5;
  f8 acc[4];
#pragma unroll
  for (int j = 0; j < 4; ++j) acc[j] = f8_zero();
  const int ar = tid >> 1;
  const int aseg = (tid & 1) << 4;
  int row = m0 + ar; if (row >= m) row = m - 1;
  const int oy = row / side;
  const int ox = row - oy * side;
  const _Float16* xb = Ximg + (size_t)b * CC * NN;
  const int pb = tid * 8;
  const int nb = n0 + (pb >> 5), cb = pb & 31;

  auto issueB = [&](int k0, int buf) {
    async_ld_b128(lds_addr(&Bs[buf][pb]), &Wt[(size_t)nb * KTOT + k0 + cb]);
  };

  const int nk = KTOT / 32;
  issueB(0, 0);
  for (int i = 0; i < nk; ++i) {
    const int buf = i & 1;
    const int k0 = i * 32;
#pragma unroll
    for (int t16 = 0; t16 < 16; ++t16) {           // im2col gather into A tile
      int t = k0 + aseg + t16;
      int ci = t / (KK * KK);
      int rem = t - ci * (KK * KK);
      int ky = rem / KK;
      int kx = rem - ky * KK;
      As[ar * 32 + aseg + t16] = xb[ci * NN + (oy + ky) * 64 + (ox + kx)];
    }
    if (i + 1 < nk) {
      issueB(k0 + 32, buf ^ 1);
      WAIT_ASYNC(0x1);
    } else {
      WAIT_ASYNC(0x0);
    }
    __syncthreads();
    h16 af = load_a_frag(&As[(wave << 4) * 32], 32);
#pragma unroll
    for (int j = 0; j < 4; ++j) {
      h16 bf = load_b_fragT(&Bs[buf][(j * 16) * 32], 32);
      acc[j] = __builtin_amdgcn_wmma_f32_16x16x32_f16(false, af, false, bf,
                                                      (short)0, acc[j], false, false);
    }
    __syncthreads();
  }
  const int lane = tid & 31;
  const int cn = lane & 15;
  const int cm = (lane >> 4) << 3;
  const int rowbase = m0 + (wave << 4) + cm;
#pragma unroll
  for (int j = 0; j < 4; ++j) {
    int col = n0 + j * 16 + cn;
    float bv = bias[col];
#pragma unroll
    for (int r = 0; r < 8; ++r) {
      int rr = rowbase + r;
      if (rr < m)
        Out[((size_t)b * m + rr) * CC + col] = acc[j][r] + bv;
    }
  }
}

// ---------------- LayerNorm + exact GELU, one wave per 192-wide row ----------------

__global__ __launch_bounds__(256) void k_ln_gelu(
    const float* __restrict__ T, const float* __restrict__ g,
    const float* __restrict__ be, _Float16* __restrict__ Out, int rows) {
  int row = blockIdx.x * 8 + (threadIdx.x >> 5);
  if (row >= rows) return;
  int lane = threadIdx.x & 31;
  const float* p = T + (size_t)row * CC;
  float v[6], s = 0.f, s2 = 0.f;
#pragma unroll
  for (int i = 0; i < 6; ++i) { v[i] = p[lane + 32 * i]; s += v[i]; s2 += v[i] * v[i]; }
#pragma unroll
  for (int off = 1; off < 32; off <<= 1) {
    s += __shfl_xor(s, off, 32);
    s2 += __shfl_xor(s2, off, 32);
  }
  float mean = s * (1.0f / 192.0f);
  float var = s2 * (1.0f / 192.0f) - mean * mean;
  float inv = rsqrtf(var + LN_EPS);
  _Float16* o = Out + (size_t)row * CC;
#pragma unroll
  for (int i = 0; i < 6; ++i) {
    int c = lane + 32 * i;
    float y = (v[i] - mean) * inv * g[c] + be[c];
    float ge = 0.5f * y * (1.0f + erff(y * 0.70710678118654752f));
    o[c] = (_Float16)ge;
  }
}

// ---------------- split KV + fused depthwise 3x3 local conv on V ----------------
// K row-major (key-major, padded to mpad keys, pad = 0).
// V transposed: VT[(b*64 + d) * mpad + key] (key-contiguous, pad = 0).

__global__ void k_split_lc(const float* __restrict__ KV,
                           const float* __restrict__ lcw, const float* __restrict__ lcb,
                           _Float16* __restrict__ Kh, _Float16* __restrict__ VT,
                           int m, int mpad, int side) {
  int idx = blockIdx.x * 256 + threadIdx.x;   // over NB*mpad*64
  int c = idx & 63;
  int t = idx >> 6;
  int pix = t % mpad;
  int b = t / mpad;
  if (pix >= m) {
    Kh[((size_t)b * mpad + pix) * 64 + c] = (_Float16)0.0f;
    VT[((size_t)b * 64 + c) * mpad + pix] = (_Float16)0.0f;
    return;
  }
  const float* kvb = KV + (size_t)b * m * 128;
  Kh[((size_t)b * mpad + pix) * 64 + c] = (_Float16)kvb[(size_t)pix * 128 + c];
  int y = pix / side, x = pix - y * side;
  float acc = lcb[c];
#pragma unroll
  for (int dy = 0; dy < 3; ++dy)
#pragma unroll
    for (int dx = 0; dx < 3; ++dx) {
      int yy = y + dy - 1, xx = x + dx - 1;
      if (yy >= 0 && yy < side && xx >= 0 && xx < side)
        acc += kvb[(size_t)(yy * side + xx) * 128 + 64 + c] * lcw[c * 9 + dy * 3 + dx];
    }
  VT[((size_t)b * 64 + c) * mpad + pix] =
      (_Float16)(kvb[(size_t)pix * 128 + 64 + c] + acc);
}

// ---------------- flash attention: wave = 16 queries x 64 dims, stream 32 keys ----------------

__global__ __launch_bounds__(128) void k_flash(
    const _Float16* __restrict__ Q, const _Float16* __restrict__ Kb,
    const _Float16* __restrict__ VT, _Float16* __restrict__ Oc,
    int m, int mpad, int hoff) {
  __shared__ __align__(16) _Float16 Pl[4][16 * 32];
  const int b = blockIdx.y;
  const int wave = threadIdx.x >> 5;
  const int lane = threadIdx.x & 31;
  const int cn = lane & 15;
  const int cm = (lane >> 4) << 3;
  const int krow0 = (lane >> 4) << 4;
  const int qr0 = blockIdx.x * 64 + wave * 16;
  const _Float16* qb = Q + ((size_t)b * NN + qr0) * CC + hoff;
  h16 aq0 = load_a_frag(qb, CC);
  h16 aq1 = load_a_frag(qb + 32, CC);
  const _Float16* kbase = Kb + (size_t)b * mpad * HDIM;
  const _Float16* vbase = VT + (size_t)b * HDIM * mpad;
  f8 o[4];
#pragma unroll
  for (int j = 0; j < 4; ++j) o[j] = f8_zero();
  float rmax[8], rsum[8];
#pragma unroll
  for (int r = 0; r < 8; ++r) { rmax[r] = -3.0e38f; rsum[r] = 0.0f; }
  _Float16* pl = &Pl[wave][0];
  for (int kb0 = 0; kb0 < m; kb0 += 32) {
    if (kb0 + 32 < m) {   // prefetch next K tile (global_prefetch_b8), uniform branch
      __builtin_prefetch(kbase + (size_t)(kb0 + 32 + lane) * HDIM, 0, 1);
    }
    f8 s[2];
    s[0] = f8_zero(); s[1] = f8_zero();
#pragma unroll
    for (int n = 0; n < 2; ++n) {               // S = Q(16x64) @ K^T(64x32)
      const _Float16* kr = kbase + (size_t)(kb0 + n * 16 + cn) * HDIM;
      h16 bf0 = join16(*(const h8*)(kr + krow0), *(const h8*)(kr + krow0 + 8));
      h16 bf1 = join16(*(const h8*)(kr + 32 + krow0), *(const h8*)(kr + 32 + krow0 + 8));
      s[n] = __builtin_amdgcn_wmma_f32_16x16x32_f16(false, aq0, false, bf0,
                                                    (short)0, s[n], false, false);
      s[n] = __builtin_amdgcn_wmma_f32_16x16x32_f16(false, aq1, false, bf1,
                                                    (short)0, s[n], false, false);
    }
    const bool mask0 = (kb0 + cn) >= m;
    const bool mask1 = (kb0 + 16 + cn) >= m;
    float mc[8];
#pragma unroll
    for (int r = 0; r < 8; ++r) {
      float a0 = mask0 ? -3.0e38f : s[0][r] * ATT_SCALE;
      float a1 = mask1 ? -3.0e38f : s[1][r] * ATT_SCALE;
      s[0][r] = a0; s[1][r] = a1;
      mc[r] = fmaxf(a0, a1);
    }
#pragma unroll
    for (int off = 1; off <= 8; off <<= 1)      // row reduce within 16-lane half
#pragma unroll
      for (int r = 0; r < 8; ++r) mc[r] = fmaxf(mc[r], __shfl_xor(mc[r], off, 32));
    float corr[8];
#pragma unroll
    for (int r = 0; r < 8; ++r) {
      float mn = fmaxf(rmax[r], mc[r]);
      corr[r] = __expf(rmax[r] - mn);
      rmax[r] = mn;
    }
#pragma unroll
    for (int j = 0; j < 4; ++j)
#pragma unroll
      for (int r = 0; r < 8; ++r) o[j][r] *= corr[r];
    float ps[8];
#pragma unroll
    for (int r = 0; r < 8; ++r) {
      float p0 = __expf(s[0][r] - rmax[r]);
      float p1 = __expf(s[1][r] - rmax[r]);
      ps[r] = p0 + p1;
      pl[(cm + r) * 32 + cn] = (_Float16)p0;     // repack P (C-layout -> A-layout)
      pl[(cm + r) * 32 + 16 + cn] = (_Float16)p1;
    }
#pragma unroll
    for (int off = 1; off <= 8; off <<= 1)
#pragma unroll
      for (int r = 0; r < 8; ++r) ps[r] += __shfl_xor(ps[r], off, 32);
#pragma unroll
    for (int r = 0; r < 8; ++r) rsum[r] = rsum[r] * corr[r] + ps[r];
    h16 ap = load_a_frag(pl, 32);
#pragma unroll
    for (int j = 0; j < 4; ++j) {                // O += P(16x32) @ V(32x64), V^T K-contig
      h16 bv = load_b_fragT(vbase + (size_t)(j * 16) * mpad + kb0, mpad);
      o[j] = __builtin_amdgcn_wmma_f32_16x16x32_f16(false, ap, false, bv,
                                                    (short)0, o[j], false, false);
    }
  }
#pragma unroll
  for (int j = 0; j < 4; ++j)
#pragma unroll
    for (int r = 0; r < 8; ++r) {
      float v = o[j][r] / rsum[r];
      Oc[((size_t)b * NN + qr0 + cm + r) * CC + hoff + j * 16 + cn] = (_Float16)v;
    }
}

// ---------------- host driver ----------------

extern "C" void kernel_launch(void* const* d_in, const int* in_sizes, int n_in,
                              void* d_out, int out_size, void* d_ws, size_t ws_size,
                              hipStream_t stream) {
  (void)in_sizes; (void)n_in; (void)out_size; (void)ws_size;
  const float* x = (const float*)d_in[0];
  const float* q_w = (const float*)d_in[1];
  const float* q_b = (const float*)d_in[2];
  const float* kv_w = (const float*)d_in[3];
  const float* kv_b = (const float*)d_in[4];
  const float* sr_w[3] = {(const float*)d_in[5], (const float*)d_in[7], (const float*)d_in[9]};
  const float* sr_b[3] = {(const float*)d_in[6], (const float*)d_in[8], (const float*)d_in[10]};
  const float* ln_g[3] = {(const float*)d_in[11], (const float*)d_in[13], (const float*)d_in[15]};
  const float* ln_b[3] = {(const float*)d_in[12], (const float*)d_in[14], (const float*)d_in[16]};
  const float* lc_w[3] = {(const float*)d_in[17], (const float*)d_in[19], (const float*)d_in[21]};
  const float* lc_b[3] = {(const float*)d_in[18], (const float*)d_in[20], (const float*)d_in[22]};
  const float* nn1_w = (const float*)d_in[23];
  const float* nn1_b = (const float*)d_in[24];
  float* out = (float*)d_out;

  char* base = (char*)d_ws;
  size_t off = 0;
  auto alloc = [&](size_t bytes) -> void* {
    void* p = base + off;
    off += (bytes + 255) & ~(size_t)255;
    return p;
  };
  const int mmaxpad = 3984;                        // (3969+15) & ~15; mpad for m = 3969 is 4000
  _Float16* xh   = (_Float16*)alloc((size_t)NB * NN * CC * 2);
  _Float16* ximg = (_Float16*)alloc((size_t)NB * NN * CC * 2);
  _Float16* wqT  = (_Float16*)alloc((size_t)192 * 192 * 2);   // N x K (native)
  _Float16* wkvT = (_Float16*)alloc((size_t)128 * 192 * 2);   // first 128 rows
  _Float16* wnnT = (_Float16*)alloc((size_t)192 * 192 * 2);
  _Float16* srT  = (_Float16*)alloc((size_t)192 * 192 * 64 * 2); // max (co x ci*8*8)
  _Float16* qh   = (_Float16*)alloc((size_t)NB * NN * CC * 2);
  float*    tf   = (float*)alloc((size_t)NB * 3969 * CC * 4);
  _Float16* tg   = (_Float16*)alloc((size_t)NB * 3969 * CC * 2);
  float*    kvf  = (float*)alloc((size_t)NB * 3969 * 128 * 4);
  _Float16* kh   = (_Float16*)alloc((size_t)NB * mmaxpad * 64 * 2);
  _Float16* vt   = (_Float16*)alloc((size_t)NB * 64 * mmaxpad * 2);
  _Float16* xcat = (_Float16*)alloc((size_t)NB * NN * CC * 2);

  k_cvt_x<<<NB * NN * CC / 256, 256, 0, stream>>>(x, xh, ximg);
  k_f2h<<<192 * 192 / 256, 256, 0, stream>>>(q_w, wqT);
  k_f2h<<<128 * 192 / 256, 256, 0, stream>>>(kv_w, wkvT);   // rows 0..127 only
  k_f2h<<<192 * 192 / 256, 256, 0, stream>>>(nn1_w, wnnT);

  // Q projection -> f16 (B,N,192)
  k_gemm<<<dim3(NB * NN / 128, 3), 256, 0, stream>>>(xh, wqT, q_b, nullptr, qh,
                                                     NB * NN, 192, 192);

  const int kks[3] = {8, 4, 2};
  const int sides[3] = {57, 61, 63};
  for (int h = 0; h < 3; ++h) {
    const int kk = kks[h], side = sides[h], m = side * side;
    const int mpad = (m + 31) & ~31;
    const int K = CC * kk * kk;
    k_f2h<<<(unsigned)((size_t)192 * K / 256), 256, 0, stream>>>(sr_w[h], srT);
    dim3 cg((unsigned)((m + 127) / 128), 3, NB);
    if (kk == 8)      k_conv_gemm<8><<<cg, 256, 0, stream>>>(ximg, srT, sr_b[h], tf, side, m);
    else if (kk == 4) k_conv_gemm<4><<<cg, 256, 0, stream>>>(ximg, srT, sr_b[h], tf, side, m);
    else              k_conv_gemm<2><<<cg, 256, 0, stream>>>(ximg, srT, sr_b[h], tf, side, m);
    const int rows = NB * m;
    k_ln_gelu<<<(unsigned)((rows + 7) / 8), 256, 0, stream>>>(tf, ln_g[h], ln_b[h], tg, rows);
    k_gemm<<<dim3((unsigned)((rows + 127) / 128), 2), 256, 0, stream>>>(
        tg, wkvT, kv_b, kvf, nullptr, rows, 192, 128);
    k_split_lc<<<(unsigned)((size_t)NB * mpad * 64 / 256), 256, 0, stream>>>(
        kvf, lc_w[h], lc_b[h], kh, vt, m, mpad, side);
    k_flash<<<dim3(NN / 64, NB), 128, 0, stream>>>(qh, kh, vt, xcat, m, mpad, h * 64);
  }

  // output projection -> f32 d_out
  k_gemm<<<dim3(NB * NN / 128, 3), 256, 0, stream>>>(xcat, wnnT, nn1_b, out, nullptr,
                                                     NB * NN, 192, 192);
}
